// GAT_x3_GATP_MH_62749472195191
// MI455X (gfx1250) — matrix operand order repaired and checked by reading, not compiled, not run on hardware
//
#include <hip/hip_runtime.h>

// ---------------------------------------------------------------------------
// GNN inference pipeline for MI455X (gfx1250, wave32, WMMA + async-LDS).
//  - one-time fp32->bf16 conversion; weights transposed to [N][K] so both
//    A and B LDS tiles have identical pair-contiguous layout -> all fragment
//    reads are ds_load_b128 (no ds_load_u16 / packing VALU in the hot loop)
//  - bf16 WMMA GEMM, double-buffered LDS tiles staged with
//    global_load_async_to_lds_b128 (ASYNCcnt), WMMA on v_wmma_f32_16x16x32_bf16
// ---------------------------------------------------------------------------

#define NN 8192
#define EE 131072
#define BB 32
#define F_IN 16928
#define DD 2048
#define HH 16
#define CC 128
#define LL 3
#define BN_EPS 0.2f

typedef __attribute__((ext_vector_type(16))) __bf16 v16bf;
typedef __attribute__((ext_vector_type(8)))  float  v8f;
union FragBF { v16bf v; unsigned u[8]; };

#if defined(__gfx1250__)
  #if __has_builtin(__builtin_amdgcn_s_wait_asynccnt)
    #define WAIT_ASYNC() __builtin_amdgcn_s_wait_asynccnt(0)
  #else
    #define WAIT_ASYNC() asm volatile("s_wait_asynccnt 0x0" ::: "memory")
  #endif
  // Per-lane 16B global -> LDS async copy. LDS aperture: generic-pointer
  // low 32 bits are the LDS byte offset (ISA 10.2).
  #define COPY16(la, ga)                                                      \
    asm volatile("global_load_async_to_lds_b128 %0, %1, off"                  \
                 :: "v"((unsigned)(unsigned long long)(la)),                  \
                    "v"((unsigned long long)(ga)) : "memory")
#else
  #define WAIT_ASYNC()
  #define COPY16(la, ga) (*(uint4*)(la) = *(const uint4*)(ga))
#endif

__device__ __forceinline__ unsigned short f2bf(float f) {
  unsigned u = __float_as_uint(f);
  u += 0x7FFFu + ((u >> 16) & 1u);           // round-to-nearest-even
  return (unsigned short)(u >> 16);
}

// Monotonic uint key for float atomic-max (handles negatives).
__device__ __forceinline__ unsigned fkey(float x) {
  unsigned u = __float_as_uint(x);
  return (u & 0x80000000u) ? ~u : (u | 0x80000000u);
}
__device__ __forceinline__ float funkey(unsigned k) {
  unsigned u = (k & 0x80000000u) ? (k & 0x7FFFFFFFu) : ~k;
  return __uint_as_float(u);
}

__device__ __forceinline__ float waveRed32(float v) {
  #pragma unroll
  for (int o = 16; o > 0; o >>= 1) v += __shfl_down(v, o, 32);
  return v;
}

// ---------------------------------------------------------------------------
// fp32 -> packed bf16 converter (8 elems / thread, 16B stores)
// ---------------------------------------------------------------------------
__global__ __launch_bounds__(256)
void cvt_f32_bf16(const float* __restrict__ src, unsigned short* __restrict__ dst,
                  long long n8) {
  const long long t = (long long)blockIdx.x * 256 + threadIdx.x;
  if (t >= n8) return;
  const float4 a = ((const float4*)src)[t * 2];
  const float4 b = ((const float4*)src)[t * 2 + 1];
  uint4 o;
  o.x = (unsigned)f2bf(a.x) | ((unsigned)f2bf(a.y) << 16);
  o.y = (unsigned)f2bf(a.z) | ((unsigned)f2bf(a.w) << 16);
  o.z = (unsigned)f2bf(b.x) | ((unsigned)f2bf(b.y) << 16);
  o.w = (unsigned)f2bf(b.z) | ((unsigned)f2bf(b.w) << 16);
  ((uint4*)dst)[t] = o;
}

// ---------------------------------------------------------------------------
// fp32 [K,Nn] -> bf16 transposed [Nn,K]. 32x32 tiles via LDS, block 256.
// grid = (Nn/32, K/32)
// ---------------------------------------------------------------------------
__global__ __launch_bounds__(256)
void cvt_transpose_bf16(const float* __restrict__ src,
                        unsigned short* __restrict__ dst, int K, int Nn) {
  __shared__ float t[32][33];
  const int tx = threadIdx.x & 31, ty = threadIdx.x >> 5;   // 32 x 8
  const int n0 = blockIdx.x * 32, k0 = blockIdx.y * 32;
  #pragma unroll
  for (int r = 0; r < 4; ++r) {
    const int row = ty + r * 8;                              // k within tile
    t[row][tx] = src[(size_t)(k0 + row) * Nn + n0 + tx];
  }
  __syncthreads();
  #pragma unroll
  for (int r = 0; r < 4; ++r) {
    const int row = ty + r * 8;                              // n within tile
    dst[(size_t)(n0 + row) * K + k0 + tx] = f2bf(t[tx][row]);
  }
}

// ---------------------------------------------------------------------------
// bf16 WMMA GEMM: out = epilogue(A[M,K] @ BT[Nn,K]^T), accum fp32.
// Block: 256 threads (8 waves); block tile 128x128, K step 32; wave 64x32.
// Both tiles are 128 rows x 32 k (stride 40 ushorts = 80B, 16B-multiple),
// staged by per-lane async global->LDS b128 copies, double-buffered.
// Requires: K % 32 == 0, Nn % 128 == 0, A padded to 128-row multiple.
// mode 0: none | 1: leaky(0.01)+BN | 2: relu+BN. C (fp32) / Cb (bf16) optional.
// ---------------------------------------------------------------------------
__global__ __launch_bounds__(256)
void gemm_bf16_wmma(const unsigned short* __restrict__ A,
                    const unsigned short* __restrict__ BT,
                    float* __restrict__ C, unsigned short* __restrict__ Cb,
                    int M, int Nn, int K, int mode,
                    const float* __restrict__ gamma, const float* __restrict__ beta,
                    const float* __restrict__ mean,  const float* __restrict__ var) {
  __shared__ unsigned short As[2][128 * 40];  // [row][k]
  __shared__ unsigned short Bs[2][128 * 40];  // [col][k]

  const int tid  = threadIdx.x;
  const int lane = tid & 31;
  const int wave = tid >> 5;
  const int wr   = wave >> 2;               // 0..1
  const int wc   = wave & 3;                // 0..3
  const int bRow = blockIdx.y * 128;
  const int bCol = blockIdx.x * 128;

  const int lrow  = lane & 15;
  const int khalf = (lane < 16) ? 0 : 8;    // A frag K offset
  const int kbB   = (lane < 16) ? 0 : 16;   // B frag K base

  // Stage one 128x32 A tile + 128x32 BT tile (8KB each) into buffer `buf`.
  // 512 x 16B transfers per tile; 2 A + 2 B per thread (per-lane async ops).
  auto stage = [&](int buf, int k0) {
    #pragma unroll
    for (int j = 0; j < 2; ++j) {
      const int q = tid + j * 256;
      const int row = q >> 2, seg = q & 3;
      const unsigned short* ga = A  + (size_t)(bRow + row) * K + k0 + seg * 8;
      const unsigned short* gb = BT + (size_t)(bCol + row) * K + k0 + seg * 8;
      unsigned short* la = &As[buf][row * 40 + seg * 8];
      unsigned short* lb = &Bs[buf][row * 40 + seg * 8];
      COPY16(la, ga);
      COPY16(lb, gb);
    }
  };

  v8f acc[4][2];
  #pragma unroll
  for (int mt = 0; mt < 4; ++mt)
    #pragma unroll
    for (int nt = 0; nt < 2; ++nt)
      #pragma unroll
      for (int r = 0; r < 8; ++r) acc[mt][nt][r] = 0.0f;

  int p = 0;
  stage(0, 0);
  for (int k0 = 0; k0 < K; k0 += 32) {
    WAIT_ASYNC();          // our wave's async copies into buf p are done
    __syncthreads();       // all waves' copies visible; prior reads of p^1 done
    if (k0 + 32 < K) stage(p ^ 1, k0 + 32);   // overlap next tile with compute

    FragBF fa[4], fb[2];
    #pragma unroll
    for (int mt = 0; mt < 4; ++mt) {
      const unsigned short* ap = &As[p][(wr * 64 + mt * 16 + lrow) * 40];
      #pragma unroll
      for (int j = 0; j < 8; ++j) {
        const int k = 2 * khalf + 2 * j;
        fa[mt].u[j] = *(const unsigned*)(ap + k);   // packed pair (k,k+1)
      }
    }
    #pragma unroll
    for (int nt = 0; nt < 2; ++nt) {
      const unsigned short* bp = &Bs[p][(wc * 32 + nt * 16 + lrow) * 40 + kbB];
      #pragma unroll
      for (int j = 0; j < 8; ++j)
        fb[nt].u[j] = *(const unsigned*)(bp + 2 * j); // 32 contiguous bytes
    }
    #pragma unroll
    for (int mt = 0; mt < 4; ++mt)
      #pragma unroll
      for (int nt = 0; nt < 2; ++nt)
        acc[mt][nt] = __builtin_amdgcn_wmma_f32_16x16x32_bf16(
            false, fa[mt].v, false, fb[nt].v, (short)0, acc[mt][nt],
            false, false);
    p ^= 1;
  }

  // -------- epilogue + store --------
  #pragma unroll
  for (int mt = 0; mt < 4; ++mt) {
    #pragma unroll
    for (int nt = 0; nt < 2; ++nt) {
      const int rbase = bRow + wr * 64 + mt * 16 + ((lane < 16) ? 0 : 8);
      const int gcol  = bCol + wc * 32 + nt * 16 + lrow;
      float g = 1.f, bta = 0.f, mu = 0.f, iv = 1.f;
      if (mode) {
        g = gamma[gcol]; bta = beta[gcol]; mu = mean[gcol];
        iv = rsqrtf(var[gcol] + BN_EPS);
      }
      #pragma unroll
      for (int r = 0; r < 8; ++r) {
        const int grow = rbase + r;
        if (grow < M) {
          float v = acc[mt][nt][r];
          if (mode == 1)      v = (v >= 0.f) ? v : 0.01f * v;
          else if (mode == 2) v = fmaxf(v, 0.f);
          if (mode) v = (v - mu) * iv * g + bta;
          if (C)  C[(size_t)grow * Nn + gcol] = v;
          if (Cb) Cb[(size_t)grow * Nn + gcol] = f2bf(v);
        }
      }
    }
  }
}

// ---------------------------------------------------------------------------
// Elementwise / attention kernels
// ---------------------------------------------------------------------------
__global__ __launch_bounds__(256)
void count_nodes(const int* __restrict__ batch, int* __restrict__ cnt) {
  const int n = blockIdx.x * blockDim.x + threadIdx.x;
  if (n < NN) atomicAdd(&cnt[batch[n]], 1);
}

// a_s[n,h] = sum_c hh[n,h,c]*att_s[h,c]; a_d likewise. One wave per (n,h).
__global__ __launch_bounds__(256)
void head_dots(const float* __restrict__ hh, const float* __restrict__ att_s,
               const float* __restrict__ att_d, float* __restrict__ a_s,
               float* __restrict__ a_d) {
  const int gt = blockIdx.x * blockDim.x + threadIdx.x;
  const int wid = gt >> 5, lane = gt & 31;
  if (wid >= NN * HH) return;
  const int n = wid >> 4, h = wid & 15;
  const int c = lane * 4;
  const float4 hv = *(const float4*)(hh + (size_t)n * DD + h * CC + c);
  const float4 sv = *(const float4*)(att_s + h * CC + c);
  const float4 dv = *(const float4*)(att_d + h * CC + c);
  float vs = hv.x * sv.x + hv.y * sv.y + hv.z * sv.z + hv.w * sv.w;
  float vd = hv.x * dv.x + hv.y * dv.y + hv.z * dv.z + hv.w * dv.w;
  vs = waveRed32(vs); vd = waveRed32(vd);
  if (lane == 0) { a_s[wid] = vs; a_d[wid] = vd; }
}

__global__ __launch_bounds__(256)
void edge_max(const int* __restrict__ ei, const float* __restrict__ a_s,
              const float* __restrict__ a_d, unsigned* __restrict__ mk) {
  const int t = blockIdx.x * blockDim.x + threadIdx.x;
  if (t >= EE * HH) return;
  const int e = t >> 4, h = t & 15;
  const int src = ei[e], dst = ei[EE + e];
  float v = a_s[src * HH + h] + a_d[dst * HH + h];
  v = (v >= 0.f) ? v : 0.2f * v;
  atomicMax(&mk[dst * HH + h], fkey(v));
}

__global__ __launch_bounds__(256)
void edge_expsum(const int* __restrict__ ei, const float* __restrict__ a_s,
                 const float* __restrict__ a_d, const unsigned* __restrict__ mk,
                 float* __restrict__ wedge, float* __restrict__ ssum) {
  const int t = blockIdx.x * blockDim.x + threadIdx.x;
  if (t >= EE * HH) return;
  const int e = t >> 4, h = t & 15;
  const int src = ei[e], dst = ei[EE + e];
  float v = a_s[src * HH + h] + a_d[dst * HH + h];
  v = (v >= 0.f) ? v : 0.2f * v;
  const float w = __expf(v - funkey(mk[dst * HH + h]));
  wedge[t] = w;
  atomicAdd(&ssum[dst * HH + h], w);
}

// One wave per (edge, head): scatter hh[src,h,:]*alpha into out[dst,h,:].
__global__ __launch_bounds__(256)
void edge_scatter(const int* __restrict__ ei, const float* __restrict__ wedge,
                  const float* __restrict__ ssum, const float* __restrict__ hh,
                  float* __restrict__ outb) {
  const int gt = blockIdx.x * blockDim.x + threadIdx.x;
  const int wid = gt >> 5, lane = gt & 31;
  if (wid >= EE * HH) return;
  const int e = wid >> 4, h = wid & 15;
  const int src = ei[e], dst = ei[EE + e];
  const float alpha = wedge[wid] / ssum[dst * HH + h];
  const int c = lane * 4;
  const float4 hv = *(const float4*)(hh + (size_t)src * DD + h * CC + c);
  float* o = outb + (size_t)dst * DD + h * CC + c;
  atomicAdd(o + 0, hv.x * alpha);
  atomicAdd(o + 1, hv.y * alpha);
  atomicAdd(o + 2, hv.z * alpha);
  atomicAdd(o + 3, hv.w * alpha);
}

// h_bf16 = bf16( relu(out + bg) * 1/sqrt(max(cnt[batch[n]],1)) )
__global__ __launch_bounds__(256)
void layer_update(const float* __restrict__ outb, const float* __restrict__ bg,
                  const int* __restrict__ batch, const int* __restrict__ cnt,
                  unsigned short* __restrict__ hb) {
  const int t = blockIdx.x * blockDim.x + threadIdx.x;
  if (t >= NN * DD) return;
  const int n = t >> 11, j = t & (DD - 1);
  float v = fmaxf(outb[t] + bg[j], 0.f);
  const float cn = (float)cnt[batch[n]];
  hb[t] = f2bf(v * rsqrtf(fmaxf(cn, 1.f)));
}

// sc[n,h] = leaky0.2(dot(hp[n,h,:], att_p[h,:])); atomicMax per (batch,h).
__global__ __launch_bounds__(256)
void pool_scores(const float* __restrict__ hp, const float* __restrict__ att_p,
                 const int* __restrict__ batch, float* __restrict__ sc,
                 unsigned* __restrict__ bmax) {
  const int gt = blockIdx.x * blockDim.x + threadIdx.x;
  const int wid = gt >> 5, lane = gt & 31;
  if (wid >= NN * HH) return;
  const int n = wid >> 4, h = wid & 15;
  const int c = lane * 4;
  const float4 hv = *(const float4*)(hp + (size_t)n * DD + h * CC + c);
  const float4 av = *(const float4*)(att_p + h * CC + c);
  float v = hv.x * av.x + hv.y * av.y + hv.z * av.z + hv.w * av.w;
  v = waveRed32(v);
  if (lane == 0) {
    v = (v >= 0.f) ? v : 0.2f * v;
    sc[wid] = v;
    atomicMax(&bmax[batch[n] * HH + h], fkey(v));
  }
}

__global__ __launch_bounds__(256)
void pool_expsum(const float* __restrict__ sc, const unsigned* __restrict__ bmax,
                 const int* __restrict__ batch, float* __restrict__ wnode,
                 float* __restrict__ bsum) {
  const int t = blockIdx.x * blockDim.x + threadIdx.x;
  if (t >= NN * HH) return;
  const int n = t >> 4, h = t & 15;
  const int b = batch[n];
  const float w = __expf(sc[t] - funkey(bmax[b * HH + h]));
  wnode[t] = w;
  atomicAdd(&bsum[b * HH + h], w);
}

__global__ __launch_bounds__(256)
void pool_agg(const float* __restrict__ hp, const float* __restrict__ wnode,
              const float* __restrict__ bsum, const int* __restrict__ batch,
              float* __restrict__ pooled) {
  const int gt = blockIdx.x * blockDim.x + threadIdx.x;
  const int wid = gt >> 5, lane = gt & 31;
  if (wid >= NN * HH) return;
  const int n = wid >> 4, h = wid & 15;
  const int b = batch[n];
  const float alpha = wnode[wid] / bsum[b * HH + h];
  const int c = lane * 4;
  const float4 hv = *(const float4*)(hp + (size_t)n * DD + h * CC + c);
  float* o = pooled + (size_t)b * DD + h * CC + c;
  atomicAdd(o + 0, hv.x * alpha);
  atomicAdd(o + 1, hv.y * alpha);
  atomicAdd(o + 2, hv.z * alpha);
  atomicAdd(o + 3, hv.w * alpha);
}

// out[b,j] = z[b,:] @ W3[:,j] + b3[j]   (32x10, K=1024)
__global__ __launch_bounds__(64)
void final_linear(const float* __restrict__ z, const float* __restrict__ W3,
                  const float* __restrict__ b3, float* __restrict__ out) {
  const int t = blockIdx.x * blockDim.x + threadIdx.x;
  if (t >= BB * 10) return;
  const int b = t / 10, j = t % 10;
  float s = b3[j];
  for (int k = 0; k < 1024; ++k) s += z[b * 1024 + k] * W3[k * 10 + j];
  out[t] = s;
}

// ---------------------------------------------------------------------------
// Host orchestration
// ---------------------------------------------------------------------------
extern "C" void kernel_launch(void* const* d_in, const int* in_sizes, int n_in,
                              void* d_out, int out_size, void* d_ws, size_t ws_size,
                              hipStream_t stream) {
  const float* x       = (const float*)d_in[0];
  const int*   ei      = (const int*)d_in[1];
  const int*   batch   = (const int*)d_in[2];
  const float* W0      = (const float*)d_in[3];
  const float* bn1_g   = (const float*)d_in[4];
  const float* bn1_b   = (const float*)d_in[5];
  const float* bn1_m   = (const float*)d_in[6];
  const float* bn1_v   = (const float*)d_in[7];
  const float* Wg      = (const float*)d_in[8];
  const float* att_src = (const float*)d_in[9];
  const float* att_dst = (const float*)d_in[10];
  const float* bg      = (const float*)d_in[11];
  const float* Wp      = (const float*)d_in[12];
  const float* att_p   = (const float*)d_in[13];
  const float* W2      = (const float*)d_in[14];
  const float* bn2_g   = (const float*)d_in[15];
  const float* bn2_b   = (const float*)d_in[16];
  const float* bn2_m   = (const float*)d_in[17];
  const float* bn2_v   = (const float*)d_in[18];
  const float* W3      = (const float*)d_in[19];
  const float* b3      = (const float*)d_in[20];
  float* outp = (float*)d_out;

  char* ws = (char*)d_ws;
  size_t off = 0;
  auto alloc = [&](size_t nbytes) {
    void* p = ws + off;
    off = (off + nbytes + 255) & ~size_t(255);
    return p;
  };
  typedef unsigned short us;
  us*       x_bf   = (us*)alloc((size_t)NN * F_IN * 2);
  us*       w0_t   = (us*)alloc((size_t)F_IN * DD * 2);   // [DD, F_IN]
  us*       wg_t   = (us*)alloc((size_t)LL * DD * DD * 2);
  us*       wp_t   = (us*)alloc((size_t)DD * DD * 2);
  us*       w2_t   = (us*)alloc((size_t)DD * 1024 * 2);   // [1024, DD]
  us*       h_bf   = (us*)alloc((size_t)NN * DD * 2);
  us*       pl_bf  = (us*)alloc((size_t)128 * DD * 2);    // pooled, 128-row pad
  float*    f_hh   = (float*)alloc((size_t)NN * DD * 4);  // hh / hp
  float*    f_out  = (float*)alloc((size_t)NN * DD * 4);
  float*    f_as   = (float*)alloc((size_t)NN * HH * 4);
  float*    f_ad   = (float*)alloc((size_t)NN * HH * 4);
  unsigned* u_mk   = (unsigned*)alloc((size_t)NN * HH * 4);
  float*    f_ssum = (float*)alloc((size_t)NN * HH * 4);
  float*    f_wedg = (float*)alloc((size_t)EE * HH * 4);
  float*    f_sc   = (float*)alloc((size_t)NN * HH * 4);
  float*    f_wn   = (float*)alloc((size_t)NN * HH * 4);
  unsigned* u_bmax = (unsigned*)alloc((size_t)BB * HH * 4);
  float*    f_bsum = (float*)alloc((size_t)BB * HH * 4);
  int*      i_cnt  = (int*)alloc((size_t)BB * 4);
  float*    f_pool = (float*)alloc((size_t)BB * DD * 4);
  float*    f_z    = (float*)alloc((size_t)BB * 1024 * 4);

  const dim3 blk(256);
  auto cvt = [&](const float* s, us* d, long long n) {
    long long n8 = n / 8;
    cvt_f32_bf16<<<(unsigned)((n8 + 255) / 256), blk, 0, stream>>>(s, d, n8);
  };
  auto cvtT = [&](const float* s, us* d, int K_, int N_) {
    dim3 g(N_ / 32, K_ / 32);
    cvt_transpose_bf16<<<g, blk, 0, stream>>>(s, d, K_, N_);
  };

  // one-time bf16 conversion; weights transposed to [N][K]
  cvt(x, x_bf, (long long)NN * F_IN);
  cvtT(W0, w0_t, F_IN, DD);
  for (int i = 0; i < LL; ++i)
    cvtT(Wg + (size_t)i * DD * DD, wg_t + (size_t)i * DD * DD, DD, DD);
  cvtT(Wp, wp_t, DD, DD);
  cvtT(W2, w2_t, DD, 1024);

  // GEMM1: h_bf = bf16(BN1(leaky(x @ W0)))   [8192,16928]x[16928,2048]
  {
    dim3 grid(DD / 128, NN / 128);
    gemm_bf16_wmma<<<grid, blk, 0, stream>>>(x_bf, w0_t, nullptr, h_bf,
                                             NN, DD, F_IN, 1,
                                             bn1_g, bn1_b, bn1_m, bn1_v);
  }

  hipMemsetAsync(i_cnt, 0, BB * 4, stream);
  count_nodes<<<(NN + 255) / 256, blk, 0, stream>>>(batch, i_cnt);

  // GAT layers
  for (int i = 0; i < LL; ++i) {
    const us*    Wgi = wg_t + (size_t)i * DD * DD;
    const float* asi = att_src + (size_t)i * HH * CC;
    const float* adi = att_dst + (size_t)i * HH * CC;
    const float* bgi = bg + (size_t)i * DD;

    dim3 grid(DD / 128, NN / 128);
    gemm_bf16_wmma<<<grid, blk, 0, stream>>>(h_bf, Wgi, f_hh, nullptr,
                                             NN, DD, DD, 0,
                                             nullptr, nullptr, nullptr, nullptr);
    head_dots<<<(NN * HH * 32) / 256, blk, 0, stream>>>(f_hh, asi, adi, f_as, f_ad);

    hipMemsetAsync(u_mk,   0, (size_t)NN * HH * 4, stream);  // key 0 == -inf
    hipMemsetAsync(f_ssum, 0, (size_t)NN * HH * 4, stream);
    hipMemsetAsync(f_out,  0, (size_t)NN * DD * 4, stream);
    edge_max   <<<(EE * HH + 255) / 256, blk, 0, stream>>>(ei, f_as, f_ad, u_mk);
    edge_expsum<<<(EE * HH + 255) / 256, blk, 0, stream>>>(ei, f_as, f_ad, u_mk,
                                                           f_wedg, f_ssum);
    edge_scatter<<<((size_t)EE * HH * 32) / 256, blk, 0, stream>>>(
        ei, f_wedg, f_ssum, f_hh, f_out);
    layer_update<<<((size_t)NN * DD + 255) / 256, blk, 0, stream>>>(
        f_out, bgi, batch, i_cnt, h_bf);
  }

  // attention pooling
  {
    dim3 grid(DD / 128, NN / 128);
    gemm_bf16_wmma<<<grid, blk, 0, stream>>>(h_bf, wp_t, f_hh, nullptr,
                                             NN, DD, DD, 0,
                                             nullptr, nullptr, nullptr, nullptr);
  }
  hipMemsetAsync(u_bmax, 0, (size_t)BB * HH * 4, stream);
  hipMemsetAsync(f_bsum, 0, (size_t)BB * HH * 4, stream);
  hipMemsetAsync(f_pool, 0, (size_t)BB * DD * 4, stream);
  pool_scores<<<(NN * HH * 32) / 256, blk, 0, stream>>>(f_hh, att_p, batch,
                                                        f_sc, u_bmax);
  pool_expsum<<<(NN * HH + 255) / 256, blk, 0, stream>>>(f_sc, u_bmax, batch,
                                                         f_wn, f_bsum);
  pool_agg<<<(NN * HH * 32) / 256, blk, 0, stream>>>(f_hh, f_wn, f_bsum, batch,
                                                     f_pool);

  // pooled -> bf16, padded to 128 rows of zeros for the unguarded tile loader
  hipMemsetAsync(pl_bf, 0, (size_t)128 * DD * 2, stream);
  cvt(f_pool, pl_bf, (long long)BB * DD);

  // lin2: z = BN2(relu(pooled @ W2))   [32,2048]x[2048,1024]
  {
    dim3 grid(1024 / 128, 1);
    gemm_bf16_wmma<<<grid, blk, 0, stream>>>(pl_bf, w2_t, f_z, nullptr,
                                             BB, 1024, DD, 2,
                                             bn2_g, bn2_b, bn2_m, bn2_v);
  }

  final_linear<<<(BB * 10 + 63) / 64, dim3(64), 0, stream>>>(f_z, W3, b3, outp);
}
